// relAttention_21285857919205
// MI455X (gfx1250) — matrix operand order repaired and checked by reading, not compiled, not run on hardware
//
#include <hip/hip_runtime.h>

#define Bsz 4
#define Ssz 1024
#define Dsz 1024
#define Hsz 16
#define DKsz 64

typedef __attribute__((ext_vector_type(16))) __bf16 v16bf;
typedef __attribute__((ext_vector_type(8)))  __bf16 v8bf;
typedef __attribute__((ext_vector_type(8)))  float  v8f;
typedef __attribute__((ext_vector_type(4)))  unsigned int v4u;

__device__ __forceinline__ v8f zero8() {
  v8f z;
#pragma unroll
  for (int i = 0; i < 8; ++i) z[i] = 0.0f;
  return z;
}

// round-to-nearest-even fp32 -> bf16, two at a time, packed into one dword
__device__ __forceinline__ unsigned bf16pair(float a, float b) {
  unsigned ua = __float_as_uint(a), ub = __float_as_uint(b);
  ua += 0x7FFFu + ((ua >> 16) & 1u);
  ub += 0x7FFFu + ((ub >> 16) & 1u);
  return (ua >> 16) | (ub & 0xFFFF0000u);
}

// build a v16bf fragment from two 16-byte LDS blocks
__device__ __forceinline__ v16bf frag_ld(const __bf16* p0, const __bf16* p1) {
  v8bf a = *(const v8bf*)p0;
  v8bf b = *(const v8bf*)p1;
  v16bf r;
#pragma unroll
  for (int i = 0; i < 8; ++i) { r[i] = a[i]; r[i + 8] = b[i]; }
  return r;
}

// CDNA5 async global->LDS copy (ASYNCcnt-tracked DMA, 16B per lane)
__device__ __forceinline__ void async_ld_b128(unsigned lds_off, const void* gaddr) {
  asm volatile("global_load_async_to_lds_b128 %0, %1, off"
               :: "v"(lds_off), "v"(gaddr) : "memory");
}
__device__ __forceinline__ void wait_async() {
  asm volatile("s_wait_asynccnt 0x0" ::: "memory");
}
// low 32 bits of a generic pointer to LDS == LDS byte offset
__device__ __forceinline__ unsigned lds_off(const void* p) {
  return (unsigned)(size_t)p;
}

// ---------------------------------------------------------------------------
// GEMM: C[4096,1024] = A[4096,1024] * W^T[1024,1024] + bias
// MODE 0: A fp32 (bf16-converted on the fly), out bf16 head-split [B,H,S,DK]
// MODE 1: A bf16, out fp32 flat [B*S, D]
// 256 threads / 8 waves, 128x128 tile, K-step 32, LDS double-buffered with
// register-staged global loads (fp32->bf16 conversion prevents async copy).
// ---------------------------------------------------------------------------
template <int MODE>
__global__ void __launch_bounds__(256) gemm_kernel(const void* __restrict__ Ain,
                                                   const float* __restrict__ W,
                                                   const float* __restrict__ bias,
                                                   void* __restrict__ Out) {
  __shared__ __bf16 Al[2][128][40];
  __shared__ __bf16 Bl[2][128][40];

  const int bm = blockIdx.x >> 3;
  const int bn = blockIdx.x & 7;
  const int t = threadIdx.x;
  const int lane = t & 31, wave = t >> 5;
  const int wm = wave >> 1, wn = wave & 1;
  const int lr = lane & 15;
  const int hi = lane >> 4;
  const int kbA = hi * 8;     // A-frag K base ({0-7,16-23}/{8-15,24-31})
  const int kbB = hi * 8;     // B-frag K base ({0-7,16-23}/{8-15,24-31})

  v8f acc[2][4];
#pragma unroll
  for (int i = 0; i < 2; ++i)
#pragma unroll
    for (int j = 0; j < 4; ++j) acc[i][j] = zero8();

  const int row = t >> 1, part = t & 1;
  const int gm0 = bm * 128, gn0 = bn * 128;

  const float*  Asrc32 = (const float*)Ain  + (size_t)(gm0 + row) * Dsz + part * 16;
  const __bf16* Asrc16 = (const __bf16*)Ain + (size_t)(gm0 + row) * Dsz + part * 16;
  const float*  Bsrc   = W                  + (size_t)(gn0 + row) * Dsz + part * 16;

  float4 ar[4], br[4];
  v4u a16[2];

  auto ldA = [&](int kt) {
    if (MODE == 0) {
#pragma unroll
      for (int j = 0; j < 4; ++j) ar[j] = *(const float4*)(Asrc32 + kt + j * 4);
    } else {
      a16[0] = *(const v4u*)(Asrc16 + kt);
      a16[1] = *(const v4u*)(Asrc16 + kt + 8);
    }
  };
  auto stA = [&](int bufi) {
    if (MODE == 0) {
#pragma unroll
      for (int j = 0; j < 4; ++j) {
        *(unsigned*)&Al[bufi][row][part * 16 + j * 4]     = bf16pair(ar[j].x, ar[j].y);
        *(unsigned*)&Al[bufi][row][part * 16 + j * 4 + 2] = bf16pair(ar[j].z, ar[j].w);
      }
    } else {
      *(v4u*)&Al[bufi][row][part * 16]     = a16[0];
      *(v4u*)&Al[bufi][row][part * 16 + 8] = a16[1];
    }
  };
  auto ldB = [&](int kt) {
#pragma unroll
    for (int j = 0; j < 4; ++j) br[j] = *(const float4*)(Bsrc + kt + j * 4);
  };
  auto stB = [&](int bufi) {
#pragma unroll
    for (int j = 0; j < 4; ++j) {
      *(unsigned*)&Bl[bufi][row][part * 16 + j * 4]     = bf16pair(br[j].x, br[j].y);
      *(unsigned*)&Bl[bufi][row][part * 16 + j * 4 + 2] = bf16pair(br[j].z, br[j].w);
    }
  };

  // pipeline prologue: tile 0 into buffer 0
  ldA(0); ldB(0); stA(0); stB(0);
  __syncthreads();

  for (int kk = 0; kk < 32; ++kk) {
    const int cur = kk & 1, nxt = cur ^ 1;
    if (kk < 31) { ldA((kk + 1) * 32); ldB((kk + 1) * 32); }  // in flight during WMMA

    v16bf af[2], bfr[4];
#pragma unroll
    for (int mt = 0; mt < 2; ++mt) {
      const __bf16* p = &Al[cur][wm * 32 + mt * 16 + lr][kbA];
      af[mt] = frag_ld(p, p + 16);
    }
#pragma unroll
    for (int nt = 0; nt < 4; ++nt) {
      const __bf16* p = &Bl[cur][wn * 64 + nt * 16 + lr][kbB];
      bfr[nt] = frag_ld(p, p + 16);
    }
#pragma unroll
    for (int mt = 0; mt < 2; ++mt)
#pragma unroll
      for (int nt = 0; nt < 4; ++nt)
        acc[mt][nt] = __builtin_amdgcn_wmma_f32_16x16x32_bf16(
            false, af[mt], false, bfr[nt], (short)0, acc[mt][nt], false, false);

    if (kk < 31) { stA(nxt); stB(nxt); }
    __syncthreads();
  }

  // epilogue: bias add + store
#pragma unroll
  for (int mt = 0; mt < 2; ++mt)
#pragma unroll
    for (int nt = 0; nt < 4; ++nt) {
      const int n = gn0 + wn * 64 + nt * 16 + lr;
      const float bb = bias[n];
#pragma unroll
      for (int r = 0; r < 8; ++r) {
        const int gm = gm0 + wm * 32 + mt * 16 + r + hi * 8;
        const float val = acc[mt][nt][r] + bb;
        if (MODE == 0) {
          const int b = gm >> 10, s = gm & 1023;
          const int h = n >> 6, dk = n & 63;
          ((__bf16*)Out)[(((size_t)(b * Hsz + h) * Ssz + s) << 6) + dk] = (__bf16)val;
        } else {
          ((float*)Out)[(size_t)gm * Dsz + n] = val;
        }
      }
    }
}

// ---------------------------------------------------------------------------
// Flash attention with relative-position bias.
// Block = 8 waves, 128 query rows of one (b,h); 16 rows per wave.
// Double-buffered 64-key tiles: K tile via GLOBAL_LOAD_ASYNC_TO_LDS_B128
// (ASYNCcnt), V tile via register transpose; Q staged once (async) in LDS
// space that aliases the K/V buffers, then lives in register fragments.
// Fragment loads are batched per K-chunk so the DS pipe runs ahead of WMMA.
// ---------------------------------------------------------------------------
__global__ void __launch_bounds__(256) attn_kernel(
    const __bf16* __restrict__ qh, const __bf16* __restrict__ kh,
    const __bf16* __restrict__ vh, const float* __restrict__ relem,
    const unsigned char* __restrict__ mask, __bf16* __restrict__ xb) {
  __shared__ alignas(16) char smem[55296];
  __bf16* P0   = (__bf16*)smem;                  // [128][72]  P (wave*16+m rows)
  __bf16* K0   = (__bf16*)(smem + 18432);        // [64][72]   K buf 0
  __bf16* K1   = (__bf16*)(smem + 27648);        // [64][72]   K buf 1
  __bf16* V0   = (__bf16*)(smem + 36864);        // [64][72]   V^T buf 0
  __bf16* V1   = (__bf16*)(smem + 46080);        // [64][72]   V^T buf 1
  __bf16* Qtmp = (__bf16*)(smem + 18432);        // [128][72]  prologue only

  const int bh = blockIdx.x >> 3;   // b*H + h
  const int qt = blockIdx.x & 7;
  const int b  = bh >> 4;
  const int h  = bh & 15;
  const int q0 = qt * 128;
  const int t = threadIdx.x, lane = t & 31, wave = t >> 5;
  const int lr = lane & 15, hi = lane >> 4;
  const int kbA = hi * 8, kbB = hi * 8;
  const int m0 = q0 + wave * 16;

  // ---- Q block: async global->LDS, then lift into register fragments ----
  {
    const int row = t >> 1, half = t & 1;
    const __bf16* src = qh + ((size_t)bh * Ssz + q0 + row) * DKsz + half * 32;
    const unsigned dst = lds_off(Qtmp + row * 72 + half * 32);
#pragma unroll
    for (int j = 0; j < 4; ++j) async_ld_b128(dst + j * 16, src + j * 8);
  }
  wait_async();
  __syncthreads();

  v16bf qf[2];
#pragma unroll
  for (int c = 0; c < 2; ++c) {
    const __bf16* p = Qtmp + (wave * 16 + lr) * 72 + c * 32 + kbA;
    qf[c] = frag_ld(p, p + 16);
  }
  __syncthreads();   // all waves done with Qtmp before K/V overwrite it

  const int kvrow = t >> 2, q4 = t & 3;   // tile loader mapping (4 thr/row)

  auto issueK = [&](__bf16* kbuf, int k0) {
    const __bf16* src = kh + ((size_t)bh * Ssz + k0 + kvrow) * DKsz + q4 * 16;
    const unsigned dst = lds_off(kbuf + kvrow * 72 + q4 * 16);
    async_ld_b128(dst, src);
    async_ld_b128(dst + 16, src + 8);
  };
  auto loadV = [&](int k0, v8bf& v0, v8bf& v1) {
    const __bf16* src = vh + ((size_t)bh * Ssz + k0 + kvrow) * DKsz + q4 * 16;
    v0 = *(const v8bf*)(src);
    v1 = *(const v8bf*)(src + 8);
  };
  auto storeVT = [&](__bf16* vbuf, const v8bf& v0, const v8bf& v1) {
#pragma unroll
    for (int i = 0; i < 8; ++i) {
      vbuf[(q4 * 16 + i) * 72 + kvrow]     = v0[i];
      vbuf[(q4 * 16 + 8 + i) * 72 + kvrow] = v1[i];
    }
  };

  float mrow[8], lrow[8];
  v8f o[4];
#pragma unroll
  for (int r = 0; r < 8; ++r) { mrow[r] = -3.0e38f; lrow[r] = 0.0f; }
#pragma unroll
  for (int i = 0; i < 4; ++i) o[i] = zero8();

  // ---- pipeline prologue: tile 0 ----
  {
    v8bf v0, v1;
    issueK(K0, 0);
    loadV(0, v0, v1);
    storeVT(V0, v0, v1);
  }
  wait_async();
  __syncthreads();

  for (int ktile = 0; ktile < 16; ++ktile) {
    const int k0 = ktile * 64;
    const int cur = ktile & 1;
    const __bf16* Kc = cur ? K1 : K0;
    const __bf16* Vc = cur ? V1 : V0;
    __bf16* Kn = cur ? K0 : K1;
    __bf16* Vn = cur ? V0 : V1;

    v8bf nv0, nv1;
    if (ktile < 15) {               // next tile's traffic in flight during compute
      issueK(Kn, k0 + 64);
      loadV(k0 + 64, nv0, nv1);
    }

    // ---- scores: 16x64 per wave; batch 4 B-fragments per K-chunk, then
    //      issue 4 independent WMMAs (DS pipe runs ahead of matrix pipe) ----
    v8f sc[4];
#pragma unroll
    for (int nt = 0; nt < 4; ++nt) sc[nt] = zero8();
#pragma unroll
    for (int c = 0; c < 2; ++c) {
      v16bf kf[4];
#pragma unroll
      for (int nt = 0; nt < 4; ++nt) {
        const __bf16* p = Kc + (nt * 16 + lr) * 72 + c * 32 + kbB;
        kf[nt] = frag_ld(p, p + 16);
      }
#pragma unroll
      for (int nt = 0; nt < 4; ++nt)
        sc[nt] = __builtin_amdgcn_wmma_f32_16x16x32_bf16(false, qf[c], false, kf[nt],
                                                         (short)0, sc[nt], false, false);
    }

    // ---- scale + relative bias + mask ----
#pragma unroll
    for (int nt = 0; nt < 4; ++nt) {
      const int n = k0 + nt * 16 + lr;
#pragma unroll
      for (int r = 0; r < 8; ++r) {
        const int m = m0 + r + hi * 8;
        float s = sc[nt][r] * 0.125f;   // 1/sqrt(64)
        s += relem[((size_t)bh * Ssz + m) * Ssz + n];
        s = mask[((size_t)b * Ssz + m) * Ssz + n] ? s : -1.0e9f;
        sc[nt][r] = s;
      }
    }

    // ---- online softmax (row reductions inside 16-lane halves) ----
    float rm[8];
#pragma unroll
    for (int r = 0; r < 8; ++r) {
      float v = fmaxf(fmaxf(sc[0][r], sc[1][r]), fmaxf(sc[2][r], sc[3][r]));
#pragma unroll
      for (int off = 1; off < 16; off <<= 1)
        v = fmaxf(v, __shfl_xor(v, off, 32));
      rm[r] = v;
    }
#pragma unroll
    for (int r = 0; r < 8; ++r) {
      const float mn = fmaxf(mrow[r], rm[r]);
      const float f = __expf(mrow[r] - mn);
      mrow[r] = mn;
      lrow[r] *= f;
#pragma unroll
      for (int ot = 0; ot < 4; ++ot) o[ot][r] *= f;
    }
    float rs[8];
#pragma unroll
    for (int r = 0; r < 8; ++r) rs[r] = 0.0f;
#pragma unroll
    for (int nt = 0; nt < 4; ++nt)
#pragma unroll
      for (int r = 0; r < 8; ++r) {
        const float p = __expf(sc[nt][r] - mrow[r]);
        sc[nt][r] = p;
        rs[r] += p;
      }
#pragma unroll
    for (int r = 0; r < 8; ++r) {
#pragma unroll
      for (int off = 1; off < 16; off <<= 1)
        rs[r] += __shfl_xor(rs[r], off, 32);
      lrow[r] += rs[r];
    }

    // ---- P -> wave-private LDS (C-layout -> A-fragment reshape) ----
#pragma unroll
    for (int nt = 0; nt < 4; ++nt)
#pragma unroll
      for (int r = 0; r < 8; ++r)
        P0[(wave * 16 + r + hi * 8) * 72 + nt * 16 + lr] = (__bf16)sc[nt][r];

    // ---- P * V: batch P-fragment + 4 V-fragments per chunk, then 4 WMMAs ----
#pragma unroll
    for (int c = 0; c < 2; ++c) {
      const __bf16* pp = P0 + (wave * 16 + lr) * 72 + c * 32 + kbA;
      v16bf pf = frag_ld(pp, pp + 16);
      v16bf vf[4];
#pragma unroll
      for (int ot = 0; ot < 4; ++ot) {
        const __bf16* vp = Vc + (ot * 16 + lr) * 72 + c * 32 + kbB;
        vf[ot] = frag_ld(vp, vp + 16);
      }
#pragma unroll
      for (int ot = 0; ot < 4; ++ot)
        o[ot] = __builtin_amdgcn_wmma_f32_16x16x32_bf16(false, pf, false, vf[ot],
                                                        (short)0, o[ot], false, false);
    }

    if (ktile < 15) storeVT(Vn, nv0, nv1);
    wait_async();       // wave's async K copies for next tile are done
    __syncthreads();    // + everyone's, and all LDS stores visible
  }

  // ---- finalize: divide by l, store merged-head context [B,S,D] bf16 ----
  float inv[8];
#pragma unroll
  for (int r = 0; r < 8; ++r) inv[r] = 1.0f / lrow[r];
#pragma unroll
  for (int ot = 0; ot < 4; ++ot) {
    const int n = h * DKsz + ot * 16 + lr;
#pragma unroll
    for (int r = 0; r < 8; ++r) {
      const int qrow = m0 + r + hi * 8;
      xb[((size_t)(b * Ssz + qrow)) * Dsz + n] = (__bf16)(o[ot][r] * inv[r]);
    }
  }
}

// ---------------------------------------------------------------------------
extern "C" void kernel_launch(void* const* d_in, const int* in_sizes, int n_in,
                              void* d_out, int out_size, void* d_ws, size_t ws_size,
                              hipStream_t stream) {
  (void)in_sizes; (void)n_in; (void)out_size; (void)ws_size;
  const float* query = (const float*)d_in[0];
  const float* key   = (const float*)d_in[1];
  const float* value = (const float*)d_in[2];
  const unsigned char* mask = (const unsigned char*)d_in[3];
  const float* relem = (const float*)d_in[4];
  const float* Wq = (const float*)d_in[5];
  const float* bq = (const float*)d_in[6];
  const float* Wk = (const float*)d_in[7];
  const float* bk = (const float*)d_in[8];
  const float* Wv = (const float*)d_in[9];
  const float* bv = (const float*)d_in[10];
  const float* Wo = (const float*)d_in[11];
  const float* bo = (const float*)d_in[12];

  char* ws = (char*)d_ws;
  const size_t seg = (size_t)Bsz * Hsz * Ssz * DKsz * sizeof(__bf16);  // 8 MB
  __bf16* qh = (__bf16*)(ws);
  __bf16* kh = (__bf16*)(ws + seg);
  __bf16* vh = (__bf16*)(ws + 2 * seg);
  __bf16* xb = (__bf16*)(ws + 3 * seg);

  dim3 blk(256);
  gemm_kernel<0><<<dim3(256), blk, 0, stream>>>(query, Wq, bq, qh);
  gemm_kernel<0><<<dim3(256), blk, 0, stream>>>(key,   Wk, bk, kh);
  gemm_kernel<0><<<dim3(256), blk, 0, stream>>>(value, Wv, bv, vh);
  attn_kernel<<<dim3(512), blk, 0, stream>>>(qh, kh, vh, relem, mask, xb);
  gemm_kernel<1><<<dim3(256), blk, 0, stream>>>(xb, Wo, bo, d_out);
}
